// KGSEncoder_13795434955243
// MI455X (gfx1250) — compile-verified, matrix operand order checked by reading, not run on hardware
//
#include <hip/hip_runtime.h>
#include <hip/hip_bf16.h>
#include <math.h>

#define N_NODES 100000
#define N_REL   500
#define DIM     200
#define N_EDGES 200000
#define T_STEPS 3

#define KPAD 224   // K padded to multiple of 32 (7 chunks)
#define NPAD 208   // output cols padded to 13 tiles of 16
#define NCT  13    // col tiles
#define MROWS 32   // rows per block (two 16-row WMMA tiles)
#define GEMM_THREADS (NCT * 32)   // 13 waves, one per col tile

typedef __attribute__((ext_vector_type(16))) __bf16          v16bf;
typedef __attribute__((ext_vector_type(8)))  float           v8f;
typedef __attribute__((ext_vector_type(8)))  unsigned short  u16x8;

union Frag {
    v16bf v;
    unsigned short s[16];
};

__device__ __forceinline__ unsigned short f2bf(float f) {
    union { float f; unsigned int u; } x;
    x.f = f;
    unsigned int u = x.u;
    u += 0x7FFFu + ((u >> 16) & 1u);   // round-to-nearest-even
    return (unsigned short)(u >> 16);
}

// ---------------------------------------------------------------------------
// Zero fill
// ---------------------------------------------------------------------------
__global__ void zero_kernel(float* p, size_t n) {
    size_t i = (size_t)blockIdx.x * blockDim.x + threadIdx.x;
    if (i < n) p[i] = 0.0f;
}

// ---------------------------------------------------------------------------
// Wave-per-row L2 normalize (wave32).  rows x DIM, out may alias in.
// ---------------------------------------------------------------------------
__global__ void l2norm_rows(const float* __restrict__ in, float* __restrict__ out, int rows) {
    int wid  = (int)((blockIdx.x * (size_t)blockDim.x + threadIdx.x) >> 5);
    int lane = threadIdx.x & 31;
    if (wid >= rows) return;
    const float* p = in + (size_t)wid * DIM;
    float v[7];
    float ss = 0.0f;
#pragma unroll
    for (int i = 0; i < 7; ++i) {
        int d = lane + i * 32;
        v[i] = (d < DIM) ? p[d] : 0.0f;
        ss += v[i] * v[i];
    }
#pragma unroll
    for (int off = 16; off > 0; off >>= 1) ss += __shfl_xor(ss, off, 32);
    float inv = 1.0f / fmaxf(sqrtf(ss), 1e-12f);
    float* q = out + (size_t)wid * DIM;
#pragma unroll
    for (int i = 0; i < 7; ++i) {
        int d = lane + i * 32;
        if (d < DIM) q[d] = v[i] * inv;
    }
}

// ---------------------------------------------------------------------------
// Weight transpose + bf16 convert:  Wt[n][k] = bf16(W[k][n]), padded w/ zeros.
// Layout: NPAD rows (cols of W) x KPAD bf16 each.
// ---------------------------------------------------------------------------
__global__ void convert_wt(const float* __restrict__ W, unsigned short* __restrict__ Wt) {
    int idx = blockIdx.x * blockDim.x + threadIdx.x;
    if (idx >= NPAD * KPAD) return;
    int n = idx / KPAD;
    int k = idx - n * KPAD;
    float v = (n < DIM && k < DIM) ? W[(size_t)k * DIM + n] : 0.0f;
    Wt[(size_t)n * KPAD + k] = f2bf(v);
}

// ---------------------------------------------------------------------------
// Per-edge degree count (fp32 atomics)
// ---------------------------------------------------------------------------
__global__ void count_deg(const int* __restrict__ edges_t, float* __restrict__ deg) {
    int e = blockIdx.x * blockDim.x + threadIdx.x;
    if (e >= N_EDGES) return;
    int dst = edges_t[e * 3 + 2];
    atomicAdd(&deg[dst], 1.0f);
}

// ---------------------------------------------------------------------------
// Scatter:  agg[dst] += h[src] + r[rel]   (aggregate BEFORE the GEMM; valid
// because segment_sum(x @ W) == segment_sum(x) @ W)
// ---------------------------------------------------------------------------
__global__ void scatter_msgs(const int* __restrict__ edges_t,
                             const float* __restrict__ h,
                             const float* __restrict__ r,
                             float* __restrict__ agg) {
    size_t tid = (size_t)blockIdx.x * blockDim.x + threadIdx.x;
    if (tid >= (size_t)N_EDGES * DIM) return;
    int e = (int)(tid / DIM);
    int d = (int)(tid - (size_t)e * DIM);
    int s   = edges_t[e * 3 + 0];
    int rel = edges_t[e * 3 + 1];
    int dst = edges_t[e * 3 + 2];
    float m = h[(size_t)s * DIM + d] + r[(size_t)rel * DIM + d];
    atomicAdd(&agg[(size_t)dst * DIM + d], m);
}

// ---------------------------------------------------------------------------
// Fused dual-source GEMM via v_wmma_f32_16x16x32_bf16:
//   C = (A1 * rowscale(deg)) @ Wt1  +  A2 @ Wt2        (A1 optional)
// One block = 13 waves; each wave owns one 16-col tile and computes a
// 32-row x 16-col output slab (two 16x16 WMMA tiles, shared B fragments).
// A is staged once per block in LDS as bf16.
// ---------------------------------------------------------------------------
__global__ __launch_bounds__(GEMM_THREADS)
void rgcn_gemm(const float* __restrict__ A1, const float* __restrict__ deg,
               const unsigned short* __restrict__ Wt1,
               const float* __restrict__ A2,
               const unsigned short* __restrict__ Wt2,
               float* __restrict__ C) {
    __shared__ __align__(16) unsigned short Abuf[2][MROWS][KPAD];
    __shared__ float scale[MROWS];

    const int tid = threadIdx.x;
    const int rowBase = blockIdx.x * MROWS;        // 100000/32 = 3125 exact
    const bool hasA1 = (A1 != nullptr);

    if (tid < MROWS) {
        scale[tid] = hasA1 ? (1.0f / fmaxf(deg[rowBase + tid], 1.0f)) : 1.0f;
    }
    __syncthreads();

    // Stage A tiles (fp32 -> bf16) into LDS, zero-padded K.
    for (int idx = tid; idx < MROWS * KPAD; idx += GEMM_THREADS) {
        int row = idx / KPAD;
        int k   = idx - row * KPAD;
        float a2 = (k < DIM) ? A2[(size_t)(rowBase + row) * DIM + k] : 0.0f;
        Abuf[1][row][k] = f2bf(a2);
        if (hasA1) {
            float a1 = (k < DIM) ? A1[(size_t)(rowBase + row) * DIM + k] * scale[row] : 0.0f;
            Abuf[0][row][k] = f2bf(a1);
        }
    }
    __syncthreads();

    const int wave = tid >> 5;                     // 0..12 == col tile
    const int lane = tid & 31;
    const int col  = wave * 16 + (lane & 15);      // < 208 always
    const int kb   = (lane >> 4) * 8;              // K sub-offset per half-wave
    const int rowA = lane & 15;

    v8f acc0 = {};                                 // rows rowBase + 0..15
    v8f acc1 = {};                                 // rows rowBase + 16..31

    for (int m = 0; m < 2; ++m) {
        if (m == 0 && !hasA1) continue;
        const unsigned short* Wt   = (m == 0) ? Wt1 : Wt2;
        const unsigned short* wrow = Wt + (size_t)col * KPAD;
        __builtin_prefetch(wrow, 0, 1);            // global_prefetch_b8
#pragma unroll
        for (int k0 = 0; k0 < KPAD; k0 += 32) {
            Frag a0, a1, b;
            *(u16x8*)&b.s[0]  = *(const u16x8*)&wrow[k0 + kb];
            *(u16x8*)&b.s[8]  = *(const u16x8*)&wrow[k0 + 16 + kb];
            *(u16x8*)&a0.s[0] = *(const u16x8*)&Abuf[m][rowA][k0 + kb];
            *(u16x8*)&a0.s[8] = *(const u16x8*)&Abuf[m][rowA][k0 + 16 + kb];
            *(u16x8*)&a1.s[0] = *(const u16x8*)&Abuf[m][rowA + 16][k0 + kb];
            *(u16x8*)&a1.s[8] = *(const u16x8*)&Abuf[m][rowA + 16][k0 + 16 + kb];
            acc0 = __builtin_amdgcn_wmma_f32_16x16x32_bf16(
                false, a0.v, false, b.v, (short)0, acc0, false, false);
            acc1 = __builtin_amdgcn_wmma_f32_16x16x32_bf16(
                false, a1.v, false, b.v, (short)0, acc1, false, false);
        }
    }

    if (col < DIM) {
        const int rshift = (lane >> 4) * 8;
#pragma unroll
        for (int v = 0; v < 8; ++v) {
            int row = rowBase + v + rshift;
            C[(size_t)row * DIM + col]        = acc0[v];
            C[(size_t)(row + 16) * DIM + col] = acc1[v];
        }
    }
}

// ---------------------------------------------------------------------------
// Final per-step update (wave per row):
//   g = sigmoid(gpre + bias);  h = l2norm(g*curn + (1-g)*h)
// ---------------------------------------------------------------------------
__global__ void gate_update(const float* __restrict__ gpre,
                            const float* __restrict__ bias,
                            const float* __restrict__ curn,
                            float* __restrict__ h, int rows) {
    int wid  = (int)((blockIdx.x * (size_t)blockDim.x + threadIdx.x) >> 5);
    int lane = threadIdx.x & 31;
    if (wid >= rows) return;
    size_t base = (size_t)wid * DIM;
    float v[7];
    float ss = 0.0f;
#pragma unroll
    for (int i = 0; i < 7; ++i) {
        int d = lane + i * 32;
        if (d < DIM) {
            float g  = 1.0f / (1.0f + expf(-(gpre[base + d] + bias[d])));
            float hv = h[base + d];
            v[i] = g * curn[base + d] + (1.0f - g) * hv;
        } else {
            v[i] = 0.0f;
        }
        ss += v[i] * v[i];
    }
#pragma unroll
    for (int off = 16; off > 0; off >>= 1) ss += __shfl_xor(ss, off, 32);
    float inv = 1.0f / fmaxf(sqrtf(ss), 1e-12f);
#pragma unroll
    for (int i = 0; i < 7; ++i) {
        int d = lane + i * 32;
        if (d < DIM) h[base + d] = v[i] * inv;
    }
}

// ---------------------------------------------------------------------------
extern "C" void kernel_launch(void* const* d_in, const int* in_sizes, int n_in,
                              void* d_out, int out_size, void* d_ws, size_t ws_size,
                              hipStream_t stream) {
    const int*   edges = (const int*)  d_in[0];   // [T][E][3] int32
    const float* ent   = (const float*)d_in[1];   // [N_NODES][DIM]
    const float* rel   = (const float*)d_in[2];   // [N_REL][DIM]
    const float* Wm1   = (const float*)d_in[3];
    const float* Wl1   = (const float*)d_in[4];
    const float* Wm2   = (const float*)d_in[5];
    const float* Wl2   = (const float*)d_in[6];
    const float* Wg    = (const float*)d_in[7];
    const float* bias  = (const float*)d_in[8];

    float* h = (float*)d_out;                     // [N_NODES][DIM], working + final

    // Workspace layout (floats, then bf16 weight copies)
    float* ws = (float*)d_ws;
    const size_t ND = (size_t)N_NODES * DIM;
    float* agg  = ws;                 // ND
    float* deg  = agg  + ND;          // N_NODES
    float* tmp1 = deg  + N_NODES;     // ND
    float* tmp2 = tmp1 + ND;          // ND
    float* gpre = tmp2 + ND;          // ND
    float* rnrm = gpre + ND;          // N_REL*DIM
    unsigned short* WtBase = (unsigned short*)(rnrm + (size_t)N_REL * DIM);
    const size_t WT = (size_t)NPAD * KPAD;
    unsigned short* WtM1 = WtBase + 0 * WT;
    unsigned short* WtL1 = WtBase + 1 * WT;
    unsigned short* WtM2 = WtBase + 2 * WT;
    unsigned short* WtL2 = WtBase + 3 * WT;
    unsigned short* WtG  = WtBase + 4 * WT;

    const int B = 256;
    const int gemmBlocks = N_NODES / MROWS;       // 3125

    // h = l2norm(entity), r = l2norm(relation)
    l2norm_rows<<<(N_NODES * 32 + B - 1) / B, B, 0, stream>>>(ent, h, N_NODES);
    l2norm_rows<<<(N_REL   * 32 + B - 1) / B, B, 0, stream>>>(rel, rnrm, N_REL);

    // bf16 transposed weights
    int cw = (int)((WT + B - 1) / B);
    convert_wt<<<cw, B, 0, stream>>>(Wm1, WtM1);
    convert_wt<<<cw, B, 0, stream>>>(Wl1, WtL1);
    convert_wt<<<cw, B, 0, stream>>>(Wm2, WtM2);
    convert_wt<<<cw, B, 0, stream>>>(Wl2, WtL2);
    convert_wt<<<cw, B, 0, stream>>>(Wg,  WtG);

    for (int t = 0; t < T_STEPS; ++t) {
        const int* et = edges + (size_t)t * N_EDGES * 3;

        // ---- layer 1: agg = segsum(h[src]+r[rel]); cur = (agg/deg)@Wm1 + h@Wl1
        zero_kernel<<<(int)((ND + N_NODES + B - 1) / B), B, 0, stream>>>(agg, ND + N_NODES);
        count_deg<<<(N_EDGES + B - 1) / B, B, 0, stream>>>(et, deg);
        scatter_msgs<<<(int)(((size_t)N_EDGES * DIM + B - 1) / B), B, 0, stream>>>(et, h, rnrm, agg);
        rgcn_gemm<<<gemmBlocks, GEMM_THREADS, 0, stream>>>(agg, deg, WtM1, h, WtL1, tmp1);

        // ---- layer 2 (same edges/deg): cur2 = (agg/deg)@Wm2 + cur@Wl2
        zero_kernel<<<(int)((ND + B - 1) / B), B, 0, stream>>>(agg, ND);
        scatter_msgs<<<(int)(((size_t)N_EDGES * DIM + B - 1) / B), B, 0, stream>>>(et, tmp1, rnrm, agg);
        rgcn_gemm<<<gemmBlocks, GEMM_THREADS, 0, stream>>>(agg, deg, WtM2, tmp1, WtL2, tmp2);

        // ---- cur = l2norm(cur2)  (in place)
        l2norm_rows<<<(N_NODES * 32 + B - 1) / B, B, 0, stream>>>(tmp2, tmp2, N_NODES);

        // ---- gate = sigmoid(h @ Wg + b); h = l2norm(gate*cur + (1-gate)*h)
        rgcn_gemm<<<gemmBlocks, GEMM_THREADS, 0, stream>>>(nullptr, nullptr, nullptr, h, WtG, gpre);
        gate_update<<<(N_NODES * 32 + B - 1) / B, B, 0, stream>>>(gpre, bias, tmp2, h, N_NODES);
    }
}
